// MambaLayerPure_51745765982855
// MI455X (gfx1250) — matrix operand order repaired and checked by reading, not compiled, not run on hardware
//
#include <hip/hip_runtime.h>

// ---------------- problem constants ----------------
#define D_MODEL 1024
#define D_STATE 16
#define D_CONV  4
#define D_INNER 2048
#define BATCH   2
#define SEQ     1024
#define M_TOK   (BATCH * SEQ)      // 2048 tokens
#define N_XZ    (2 * D_INNER)      // 4096
#define NPROJ   (2 * D_STATE + 1)  // 33

typedef __attribute__((ext_vector_type(16))) __bf16 v16bf;
typedef __attribute__((ext_vector_type(8)))  float  v8f;
typedef __attribute__((ext_vector_type(8)))  __bf16 bf16x8;

__device__ __forceinline__ float silu_f(float x)     { return x / (1.f + __expf(-x)); }
__device__ __forceinline__ float softplus_f(float x) { return (x > 20.f) ? x : __logf(1.f + __expf(x)); }

// ---------------- f32 -> bf16 convert ----------------
__global__ __launch_bounds__(256) void cvt_bf16_kernel(const float* __restrict__ src,
                                                       __bf16* __restrict__ dst, int n) {
  int i = blockIdx.x * blockDim.x + threadIdx.x;
  if (i < n) dst[i] = (__bf16)src[i];
}

// ---------------- bf16 WMMA GEMM: C[M,N] = A[M,K] x B[K,N] ----------------
// A,B bf16 row-major; C f32 row-major. M%64==0, N%128==0, K%32==0.
// Block: 256 threads = 8 waves (2 in M x 4 in N). Block tile 64x128, KC=32,
// LDS double-buffered (one barrier per K-chunk, global loads overlap WMMA).
__global__ __launch_bounds__(256) void gemm_bf16_wmma(const __bf16* __restrict__ A,
                                                      const __bf16* __restrict__ B,
                                                      float* __restrict__ C,
                                                      int M, int N, int K) {
  __shared__ __align__(16) __bf16 As[2][64][40];    // row-major A tiles (padded pitch)
  __shared__ __align__(16) __bf16 BsT[2][128][40];  // B tiles stored transposed: [n][k]

  const int tid   = threadIdx.x;
  const int lane  = tid & 31;
  const int wave  = tid >> 5;
  const int waveM = wave & 1;   // 0..1  -> 32 rows each
  const int waveN = wave >> 1;  // 0..3  -> 32 cols each
  const int half  = lane >> 4;  // 0 or 1
  const int lm    = lane & 15;

  const int rowBlk = blockIdx.y * 64;
  const int colBlk = blockIdx.x * 128;

  // cooperative-load coordinates
  const int arow = tid >> 2;           // 0..63
  const int acol = (tid & 3) * 8;      // 0,8,16,24
  const int kr0  = (tid >> 4) * 2;     // 0,2,..,30  (two consecutive K rows of B)
  const int bcol = (tid & 15) * 8;     // 0..120

  // incremented global pointers (no per-iter 64-bit multiplies)
  const __bf16* aPtr  = A + (size_t)(rowBlk + arow) * K + acol;
  const __bf16* bPtr0 = B + (size_t)kr0 * N + colBlk + bcol;
  const __bf16* bPtr1 = bPtr0 + N;
  const size_t  bStep = (size_t)32 * N;

  v8f acc[2][2] = {};

  // ---- prologue: load chunk 0, stage into buffer 0 ----
  bf16x8 av  = *(const bf16x8*)aPtr;
  bf16x8 bv0 = *(const bf16x8*)bPtr0;
  bf16x8 bv1 = *(const bf16x8*)bPtr1;

  *(bf16x8*)&As[0][arow][acol] = av;
#pragma unroll
  for (int j = 0; j < 8; ++j) {
    unsigned int p = (unsigned int)__builtin_bit_cast(unsigned short, bv0[j]) |
                     ((unsigned int)__builtin_bit_cast(unsigned short, bv1[j]) << 16);
    *(unsigned int*)&BsT[0][bcol + j][kr0] = p;  // kr0 even -> 4B aligned
  }
  __syncthreads();

  const int nch = K >> 5;
  for (int c = 0; c < nch; ++c) {
    const int cur = c & 1;
    const bool more = (c + 1) < nch;

    // ---- issue next chunk's global loads early (overlap with WMMA) ----
    if (more) {
      aPtr  += 32;
      bPtr0 += bStep;
      bPtr1 += bStep;
      av  = *(const bf16x8*)aPtr;
      bv0 = *(const bf16x8*)bPtr0;
      bv1 = *(const bf16x8*)bPtr1;
      __builtin_prefetch(aPtr + 32, 0, 3);          // global_prefetch_b8, chunk c+2
      __builtin_prefetch(bPtr0 + bStep, 0, 3);
    }

    // ---- fragments from current buffer ----
    // A (16-bit A 16x32): lanes 0-15 read K 0..15, lanes 16-31 read K 16..31 (matches B)
    v16bf afrag[2];
#pragma unroll
    for (int mt = 0; mt < 2; ++mt) {
      const int r = waveM * 32 + mt * 16 + lm;
#pragma unroll
      for (int v = 0; v < 8; ++v) {
        const int k = 2 * v + half * 16;
        afrag[mt][2 * v]     = As[cur][r][k];
        afrag[mt][2 * v + 1] = As[cur][r][k + 1];
      }
    }
    // B (32x16): contiguous k like A: lanes 0-15 K=0..15, lanes 16-31 K=16..31, N=lane&15
    v16bf bfrag[2];
#pragma unroll
    for (int nt = 0; nt < 2; ++nt) {
      const int cidx = waveN * 32 + nt * 16 + lm;
      const int kb   = half * 16;
#pragma unroll
      for (int v = 0; v < 8; ++v) {
        bfrag[nt][2 * v]     = BsT[cur][cidx][kb + 2 * v];
        bfrag[nt][2 * v + 1] = BsT[cur][cidx][kb + 2 * v + 1];
      }
    }

#pragma unroll
    for (int mt = 0; mt < 2; ++mt)
#pragma unroll
      for (int nt = 0; nt < 2; ++nt)
        acc[mt][nt] = __builtin_amdgcn_wmma_f32_16x16x32_bf16(
            false, afrag[mt], false, bfrag[nt], (short)0, acc[mt][nt], false, false);

    // ---- stage next chunk into the other buffer, single barrier per iter ----
    if (more) {
      const int nxt = cur ^ 1;
      *(bf16x8*)&As[nxt][arow][acol] = av;
#pragma unroll
      for (int j = 0; j < 8; ++j) {
        unsigned int p = (unsigned int)__builtin_bit_cast(unsigned short, bv0[j]) |
                         ((unsigned int)__builtin_bit_cast(unsigned short, bv1[j]) << 16);
        *(unsigned int*)&BsT[nxt][bcol + j][kr0] = p;
      }
      __syncthreads();
    }
  }

  // C layout: VGPR i holds M = i (+8 for lanes 16-31), N = lane&15
#pragma unroll
  for (int mt = 0; mt < 2; ++mt) {
#pragma unroll
    for (int nt = 0; nt < 2; ++nt) {
      const int col   = colBlk + waveN * 32 + nt * 16 + lm;
      const int rbase = rowBlk + waveM * 32 + mt * 16 + half * 8;
#pragma unroll
      for (int i = 0; i < 8; ++i)
        C[(size_t)(rbase + i) * N + col] = acc[mt][nt][i];
    }
  }
}

// ---------------- depthwise causal conv(k=4) + bias + SiLU ----------------
// reads x-branch half of xz (cols 0..2047), writes x_act[m][d]
__global__ __launch_bounds__(256) void conv_silu_kernel(const float* __restrict__ xz,
                                                        const float* __restrict__ cw,
                                                        const float* __restrict__ cb,
                                                        float* __restrict__ x_act) {
  const int idx = blockIdx.x * blockDim.x + threadIdx.x;
  if (idx >= BATCH * SEQ * D_INNER) return;
  const int d  = idx & (D_INNER - 1);
  const int mt = idx >> 11;          // token index m = b*SEQ + t
  const int t  = mt & (SEQ - 1);
  const int b  = mt >> 10;
  float acc = cb[d];
#pragma unroll
  for (int j = 0; j < D_CONV; ++j) {
    const int tt = t - (D_CONV - 1) + j;
    if (tt >= 0) acc += xz[(size_t)(b * SEQ + tt) * N_XZ + d] * cw[d * D_CONV + j];
  }
  x_act[(size_t)mt * D_INNER + d] = silu_f(acc);
}

// ---------------- x_proj: x_ssm[m][j] = sum_d x_act[m][d] * W[d][j], j<33 ----------------
__global__ __launch_bounds__(128) void xproj_kernel(const float* __restrict__ x_act,
                                                    const float* __restrict__ W,
                                                    float* __restrict__ x_ssm) {
  __shared__ float row[D_INNER];
  const int m   = blockIdx.x;
  const int tid = threadIdx.x;
  for (int i = tid; i < D_INNER; i += 128) row[i] = x_act[(size_t)m * D_INNER + i];
  __syncthreads();
  const int lane = tid & 31, wave = tid >> 5;
  for (int j = wave; j < NPROJ; j += 4) {
    float s = 0.f;
    for (int i = lane; i < D_INNER; i += 32) s += row[i] * W[i * NPROJ + j];
#pragma unroll
    for (int off = 16; off > 0; off >>= 1) s += __shfl_xor(s, off, 32);
    if (lane == 0) x_ssm[(size_t)m * NPROJ + j] = s;
  }
}

// ---------------- selective scan (+ D*x, * SiLU(z), -> bf16 for GEMM2) ----------------
// one channel per lane; 64 channels per block; 64 blocks (2 batches x 32 groups)
__global__ __launch_bounds__(64) void scan_kernel(const float* __restrict__ x_ssm,
                                                  const float* __restrict__ x_act,
                                                  const float* __restrict__ xz,
                                                  const float* __restrict__ dt_w,
                                                  const float* __restrict__ dt_b,
                                                  const float* __restrict__ A_log,
                                                  const float* __restrict__ D_param,
                                                  __bf16* __restrict__ yb) {
  __shared__ float st[64][NPROJ + 1];  // 64 timesteps of [B(16) C(16) dt(1)]
  const int tid = threadIdx.x;
  const int b   = blockIdx.x >> 5;
  const int d   = (blockIdx.x & 31) * 64 + tid;

  float Ac[D_STATE];
#pragma unroll
  for (int s = 0; s < D_STATE; ++s) Ac[s] = -__expf(A_log[d * D_STATE + s]);
  const float wdt = dt_w[d];
  const float bdt = dt_b[d];
  const float Dp  = D_param[d];
  float h[D_STATE] = {};

  for (int tc = 0; tc < SEQ; tc += 64) {
    const int mb = b * SEQ + tc;
    for (int i = tid; i < 64 * NPROJ; i += 64) {
      const int r = i / NPROJ, c = i - r * NPROJ;
      st[r][c] = x_ssm[(size_t)(mb + r) * NPROJ + c];
    }
    __syncthreads();
    for (int tt = 0; tt < 64; ++tt) {
      const size_t m  = (size_t)(mb + tt);
      const float dt  = softplus_f(st[tt][32] * wdt + bdt);
      const float xa  = x_act[m * D_INNER + d];
      const float zz  = xz[m * N_XZ + D_INNER + d];
      const float bx  = dt * xa;
      float y = 0.f;
#pragma unroll
      for (int s = 0; s < D_STATE; ++s) {
        const float a = __expf(dt * Ac[s]);
        h[s] = a * h[s] + bx * st[tt][s];
        y += h[s] * st[tt][D_STATE + s];
      }
      y = (y + Dp * xa) * silu_f(zz);
      yb[m * D_INNER + d] = (__bf16)y;
    }
    __syncthreads();
  }
}

// ---------------- launch ----------------
extern "C" void kernel_launch(void* const* d_in, const int* in_sizes, int n_in,
                              void* d_out, int out_size, void* d_ws, size_t ws_size,
                              hipStream_t stream) {
  const float* x    = (const float*)d_in[0];  // (B,T,1024)
  const float* w1   = (const float*)d_in[1];  // (1024,4096)
  const float* cw   = (const float*)d_in[2];  // (2048,1,4)
  const float* cb   = (const float*)d_in[3];  // (2048,)
  const float* xpw  = (const float*)d_in[4];  // (2048,33)
  const float* dtw  = (const float*)d_in[5];  // (1,2048)
  const float* dtb  = (const float*)d_in[6];  // (2048,)
  const float* Alog = (const float*)d_in[7];  // (2048,16)
  const float* Dpar = (const float*)d_in[8];  // (2048,)
  const float* w2   = (const float*)d_in[9];  // (2048,1024)
  float* out = (float*)d_out;                 // (B,T,1024) f32

  char* ws = (char*)d_ws;
  size_t off = 0;
  auto alloc = [&](size_t bytes) -> void* {
    void* p = ws + off;
    off = (off + bytes + 255) & ~(size_t)255;
    return p;
  };
  __bf16* xb    = (__bf16*)alloc((size_t)M_TOK * D_MODEL * sizeof(__bf16));   // x in bf16
  __bf16* w1b   = (__bf16*)alloc((size_t)D_MODEL * N_XZ * sizeof(__bf16));    // in_proj_w bf16
  __bf16* w2b   = (__bf16*)alloc((size_t)D_INNER * D_MODEL * sizeof(__bf16)); // c_proj_w bf16
  float*  xz    = (float*)alloc((size_t)M_TOK * N_XZ * sizeof(float));        // in_proj out
  float*  x_act = (float*)alloc((size_t)M_TOK * D_INNER * sizeof(float));     // conv+silu
  float*  x_ssm = (float*)alloc((size_t)M_TOK * NPROJ * sizeof(float));       // B,C,dt
  __bf16* yb    = (__bf16*)alloc((size_t)M_TOK * D_INNER * sizeof(__bf16));   // gated y, bf16

  // 1) convert to bf16
  {
    int n;
    n = M_TOK * D_MODEL;
    cvt_bf16_kernel<<<(n + 255) / 256, 256, 0, stream>>>(x, xb, n);
    n = D_MODEL * N_XZ;
    cvt_bf16_kernel<<<(n + 255) / 256, 256, 0, stream>>>(w1, w1b, n);
    n = D_INNER * D_MODEL;
    cvt_bf16_kernel<<<(n + 255) / 256, 256, 0, stream>>>(w2, w2b, n);
  }

  // 2) in_proj GEMM: (2048x1024) x (1024x4096) -> xz
  gemm_bf16_wmma<<<dim3(N_XZ / 128, M_TOK / 64), 256, 0, stream>>>(xb, w1b, xz,
                                                                   M_TOK, N_XZ, D_MODEL);

  // 3) depthwise conv + bias + SiLU
  conv_silu_kernel<<<(BATCH * SEQ * D_INNER) / 256, 256, 0, stream>>>(xz, cw, cb, x_act);

  // 4) x_proj (thin N=33)
  xproj_kernel<<<M_TOK, 128, 0, stream>>>(x_act, xpw, x_ssm);

  // 5) selective scan + gate, emit bf16 A-operand for out_proj
  scan_kernel<<<64, 64, 0, stream>>>(x_ssm, x_act, xz, dtw, dtb, Alog, Dpar, yb);

  // 6) out_proj GEMM: (2048x2048) x (2048x1024) -> out (f32)
  gemm_bf16_wmma<<<dim3(D_MODEL / 128, M_TOK / 64), 256, 0, stream>>>(yb, w2b, out,
                                                                      M_TOK, D_MODEL, D_INNER);
}
